// MoE_83373905150510
// MI455X (gfx1250) — compile-verified, matrix operand order checked by reading, not run on hardware
//
#include <hip/hip_runtime.h>

#define E_    8
#define TOPK_ 2
#define H_    2048
#define I_    1024
#define T_    8192

typedef __attribute__((ext_vector_type(16))) __bf16 v16bf;
typedef __attribute__((ext_vector_type(8)))  float  v8f;

// ---------- helpers ----------------------------------------------------------

__device__ __forceinline__ unsigned short f2bf(float f) {
  unsigned int u = __builtin_bit_cast(unsigned int, f);
  u += 0x7FFFu + ((u >> 16) & 1u);          // round-to-nearest-even
  return (unsigned short)(u >> 16);
}

// Load one 16x32 bf16 WMMA operand fragment. Per ISA (CDNA5 ISA):
// lanes 0-15 hold row r=lane, K = {koff..koff+7} U {16+koff..16+koff+7} with
// koff=0; lanes 16-31 the same rows with koff=8. Caller passes the per-lane
// pointer to element [row, kbase+koff]; the two 8-element strips are 16B each.
__device__ __forceinline__ v16bf frag_ld(const unsigned short* p) {
  union { uint4 q[2]; v16bf v; } u;
  u.q[0] = *(const uint4*)(p);        // K koff .. koff+7      -> elems 0..7
  u.q[1] = *(const uint4*)(p + 16);   // K 16+koff .. 16+koff+7-> elems 8..15
  return u.v;
}

// ---------- kernels ----------------------------------------------------------

__global__ __launch_bounds__(256) void k_cvt(const float* __restrict__ in,
                                             unsigned short* __restrict__ out,
                                             long long n) {
  long long i = (long long)blockIdx.x * blockDim.x + threadIdx.x;
  long long s = (long long)gridDim.x * blockDim.x;
  for (; i < n; i += s) out[i] = f2bf(in[i]);
}

__global__ __launch_bounds__(256) void k_router(
    const float* __restrict__ x, const float* __restrict__ wg,
    float* __restrict__ logits, int* __restrict__ counts,
    int* __restrict__ tke, float* __restrict__ tkw) {
  const int lane = threadIdx.x & 31;
  const int t = blockIdx.x * 8 + (threadIdx.x >> 5);   // one wave32 per token
  float acc[E_];
#pragma unroll
  for (int e = 0; e < E_; ++e) acc[e] = 0.0f;
  const float* xr = x + (size_t)t * H_;
  for (int h = lane; h < H_; h += 32) {
    float xv = xr[h];
#pragma unroll
    for (int e = 0; e < E_; ++e) acc[e] += xv * wg[e * H_ + h];
  }
#pragma unroll
  for (int e = 0; e < E_; ++e)
    for (int off = 16; off > 0; off >>= 1)
      acc[e] += __shfl_xor(acc[e], off, 32);
  if (lane == 0) {
#pragma unroll
    for (int e = 0; e < E_; ++e) logits[(size_t)t * E_ + e] = acc[e];
    int e1 = 0;
#pragma unroll
    for (int e = 1; e < E_; ++e) if (acc[e] > acc[e1]) e1 = e;   // first on ties
    int e2 = (e1 == 0) ? 1 : 0;
#pragma unroll
    for (int e = 0; e < E_; ++e) if (e != e1 && acc[e] > acc[e2]) e2 = e;
    float w1 = 1.0f / (1.0f + __expf(acc[e2] - acc[e1]));        // 2-way softmax
    tke[2 * t] = e1; tke[2 * t + 1] = e2;
    tkw[2 * t] = w1; tkw[2 * t + 1] = 1.0f - w1;
    atomicAdd(&counts[e1], 1);
    atomicAdd(&counts[e2], 1);
  }
}

__global__ void k_scan(const int* __restrict__ counts, int* __restrict__ seg_off) {
  if (threadIdx.x == 0) {
    int s = 0;
    for (int e = 0; e < E_; ++e) { seg_off[e] = s; s += counts[e]; }
  }
}

__global__ __launch_bounds__(256) void k_assign(
    const int* __restrict__ tke, const float* __restrict__ tkw,
    const int* __restrict__ seg_off, int* __restrict__ cursors,
    int* __restrict__ list, float* __restrict__ wts) {
  int t = blockIdx.x * 256 + threadIdx.x;
  if (t >= T_) return;
#pragma unroll
  for (int k = 0; k < TOPK_; ++k) {
    int e = tke[2 * t + k];
    int pos = seg_off[e] + atomicAdd(&cursors[e], 1);
    list[pos] = t;
    wts[pos]  = tkw[2 * t + k];
  }
}

// Fused expert FFN: block = 32 gathered tokens of one expert, 8 wave32s.
// Phase 1: glu[32 x 1024] (silu(x Wa^T) * (x Wb^T)) * combine_wt -> 64KB LDS.
// Phase 2: y += glu @ Wproj^T, A-fragments from LDS, atomic scatter to y.
__global__ __launch_bounds__(256) void k_moe(
    const unsigned short* __restrict__ xbf,
    const unsigned short* __restrict__ wfc,
    const unsigned short* __restrict__ wpj,
    const int* __restrict__ counts, const int* __restrict__ seg_off,
    const int* __restrict__ list, const float* __restrict__ wts,
    float* __restrict__ y) {
  __shared__ unsigned short glu_s[32 * I_];   // 64 KB (LDS is 320KB/WGP)
  __shared__ int   tok_s[32];
  __shared__ float wt_s[32];

  const int e   = blockIdx.x >> 8;            // 256 M-tiles per expert
  const int mt  = blockIdx.x & 255;
  const int cnt = counts[e];
  if (mt * 32 >= cnt) return;                 // uniform early-exit
  const int base = seg_off[e];

  const int tid = threadIdx.x;
  if (tid < 32) {
    int r = mt * 32 + tid;
    bool v = r < cnt;                         // pad rows: token 0, weight 0
    tok_s[tid] = v ? list[base + r] : 0;
    wt_s[tid]  = v ? wts[base + r] : 0.0f;
  }
  __syncthreads();

  const int w      = tid >> 5;
  const int lane   = tid & 31;
  const int r16    = lane & 15;
  const int koff   = (lane >> 4) << 3;        // 0 or 8
  const int hiMrow = (lane >> 4) << 3;        // C-frag: lanes 16-31 -> M+8

  const unsigned short* wfc_e = wfc + (size_t)e * (2 * I_) * H_;
  const unsigned short* wpj_e = wpj + (size_t)e * H_ * I_;
  const unsigned short* pA0 = xbf + (size_t)tok_s[r16]      * H_;
  const unsigned short* pA1 = xbf + (size_t)tok_s[16 + r16] * H_;

  // ---- phase 1: fc1 + GLU --------------------------------------------------
  for (int nt = 0; nt < 8; ++nt) {            // wave covers 128 glu columns
    const int nb = w * 128 + nt * 16;
    const unsigned short* pBa = wfc_e + (size_t)(nb + r16)      * H_;
    const unsigned short* pBb = wfc_e + (size_t)(I_ + nb + r16) * H_;
    v8f aca0 = {}, aca1 = {}, acb0 = {}, acb1 = {};
    for (int kk = 0; kk < H_; kk += 32) {
      __builtin_prefetch(pBa + kk + 512, 0, 0);
      v16bf A0 = frag_ld(pA0 + kk + koff);
      v16bf A1 = frag_ld(pA1 + kk + koff);
      v16bf Ba = frag_ld(pBa + kk + koff);
      v16bf Bb = frag_ld(pBb + kk + koff);
      aca0 = __builtin_amdgcn_wmma_f32_16x16x32_bf16(false, A0, false, Ba, (short)0, aca0, false, false);
      aca1 = __builtin_amdgcn_wmma_f32_16x16x32_bf16(false, A1, false, Ba, (short)0, aca1, false, false);
      acb0 = __builtin_amdgcn_wmma_f32_16x16x32_bf16(false, A0, false, Bb, (short)0, acb0, false, false);
      acb1 = __builtin_amdgcn_wmma_f32_16x16x32_bf16(false, A1, false, Bb, (short)0, acb1, false, false);
    }
#pragma unroll
    for (int v = 0; v < 8; ++v) {
      {
        int row = v + hiMrow;                 // M-tile 0: rows 0..15
        float a = aca0[v], b = acb0[v];
        float g = (a / (1.0f + __expf(-a))) * b * wt_s[row];
        glu_s[row * I_ + nb + r16] = f2bf(g);
      }
      {
        int row = 16 + v + hiMrow;            // M-tile 1: rows 16..31
        float a = aca1[v], b = acb1[v];
        float g = (a / (1.0f + __expf(-a))) * b * wt_s[row];
        glu_s[row * I_ + nb + r16] = f2bf(g);
      }
    }
  }
  __syncthreads();

  // ---- phase 2: down-proj + scatter ---------------------------------------
  for (int ht = 0; ht < 16; ++ht) {           // wave covers 256 H columns
    const int hb = w * 256 + ht * 16;
    const unsigned short* pB  = wpj_e + (size_t)(hb + r16) * I_;
    const unsigned short* lA0 = &glu_s[(size_t)r16        * I_];
    const unsigned short* lA1 = &glu_s[(size_t)(16 + r16) * I_];
    v8f ac0 = {}, ac1 = {};
    for (int kk = 0; kk < I_; kk += 32) {
      v16bf B  = frag_ld(pB  + kk + koff);
      v16bf A0 = frag_ld(lA0 + kk + koff);    // ds_load_b128 from LDS
      v16bf A1 = frag_ld(lA1 + kk + koff);
      ac0 = __builtin_amdgcn_wmma_f32_16x16x32_bf16(false, A0, false, B, (short)0, ac0, false, false);
      ac1 = __builtin_amdgcn_wmma_f32_16x16x32_bf16(false, A1, false, B, (short)0, ac1, false, false);
    }
    const int col = hb + r16;
#pragma unroll
    for (int v = 0; v < 8; ++v) {
      int row0 = v + hiMrow;
      int row1 = 16 + row0;
      atomicAdd(&y[(size_t)tok_s[row0] * H_ + col], ac0[v]);
      atomicAdd(&y[(size_t)tok_s[row1] * H_ + col], ac1[v]);
    }
  }
}

// ---------- host launcher ----------------------------------------------------

extern "C" void kernel_launch(void* const* d_in, const int* in_sizes, int n_in,
                              void* d_out, int out_size, void* d_ws, size_t ws_size,
                              hipStream_t stream) {
  (void)in_sizes; (void)n_in; (void)out_size; (void)ws_size;
  const float* x   = (const float*)d_in[0];
  const float* wg  = (const float*)d_in[1];
  const float* wfc = (const float*)d_in[2];
  const float* wpj = (const float*)d_in[3];

  float* out    = (float*)d_out;
  float* y      = out;                        // [T, H]
  float* logits = out + (size_t)T_ * H_;      // [T, E]

  char* p = (char*)d_ws;
  unsigned short* xbf  = (unsigned short*)p; p += (size_t)T_ * H_ * 2;
  unsigned short* wfcb = (unsigned short*)p; p += (size_t)E_ * 2 * I_ * H_ * 2;
  unsigned short* wpjb = (unsigned short*)p; p += (size_t)E_ * H_ * I_ * 2;
  int*   counts  = (int*)p;   p += E_ * sizeof(int);
  int*   cursors = (int*)p;   p += E_ * sizeof(int);
  int*   seg_off = (int*)p;   p += E_ * sizeof(int);
  int*   tke  = (int*)p;      p += (size_t)T_ * 2 * sizeof(int);
  float* tkw  = (float*)p;    p += (size_t)T_ * 2 * sizeof(float);
  int*   list = (int*)p;      p += (size_t)T_ * 2 * sizeof(int);
  float* wts  = (float*)p;    p += (size_t)T_ * 2 * sizeof(float);

  hipMemsetAsync(y, 0, (size_t)T_ * H_ * sizeof(float), stream);
  hipMemsetAsync(counts, 0, 3 * E_ * sizeof(int), stream);   // counts+cursors+seg_off

  k_cvt<<<2048, 256, 0, stream>>>(x,   xbf,  (long long)T_ * H_);
  k_cvt<<<4096, 256, 0, stream>>>(wfc, wfcb, (long long)E_ * 2 * I_ * H_);
  k_cvt<<<4096, 256, 0, stream>>>(wpj, wpjb, (long long)E_ * H_ * I_);

  k_router<<<T_ / 8, 256, 0, stream>>>(x, wg, logits, counts, tke, tkw);
  k_scan<<<1, 32, 0, stream>>>(counts, seg_off);
  k_assign<<<T_ / 256, 256, 0, stream>>>(tke, tkw, seg_off, cursors, list, wts);

  k_moe<<<E_ * 256, 256, 0, stream>>>(xbf, wfcb, wpjb, counts, seg_off, list, wts, y);
}
